// Encoderlayer_31267361915166
// MI455X (gfx1250) — matrix operand order repaired and checked by reading, not compiled, not run on hardware
//
#include <hip/hip_runtime.h>
#include <hip/hip_bf16.h>

// ---------------------------------------------------------------------------
// MTGNN forward for MI455X (gfx1250, wave32, WMMA).
// Heavy op: mixprop hop  h' = a*x + (1-a) * A_hat(512x512) @ H(512 x M)
// bf16 WMMA GEMM, f32 accumulate. H operand in K-blocked bf16 layout
// (Hb[k>>5][m][k&31]) so all fragments load via b128. Each wave computes a
// 16x64 strip: 1 A-frag reused across 4 B-frags -> 2.5 b128 loads per WMMA.
// ---------------------------------------------------------------------------

typedef __attribute__((ext_vector_type(16))) __bf16 bf16x16;
typedef __attribute__((ext_vector_type(8)))  __bf16 bf16x8;
typedef __attribute__((ext_vector_type(8)))  float  f32x8;

#define BB   64
#define CR   32
#define CS   64
#define NN   512
#define NDIM 40
#define TOPK 40
#define EPSF 1e-5f
#define PROP_A 0.05f

// ---------------------------------------------------------------- small utils
__global__ void k_copy(const float* __restrict__ src, float* __restrict__ dst, int n) {
    int i = blockIdx.x * blockDim.x + threadIdx.x;
    if (i < n) dst[i] = src[i];
}

// pad time dim: [B,C,N,12] -> [B,C,N,19] (7 leading zeros)
__global__ void k_pad(const float* __restrict__ x, float* __restrict__ y, int n19) {
    int i = blockIdx.x * blockDim.x + threadIdx.x;
    if (i >= n19) return;
    int t = i % 19, r = i / 19;
    y[i] = (t < 7) ? 0.0f : x[r * 12 + (t - 7)];
}

// ---------------------------------------------------------------- graph build
__global__ void k_mlin(const float* __restrict__ e, const float* __restrict__ w,
                       const float* __restrict__ b, float* __restrict__ m) {
    int i = blockIdx.x * blockDim.x + threadIdx.x;
    if (i >= NN * NDIM) return;
    int row = i / NDIM, o = i % NDIM;
    float acc = b[o];
    for (int k = 0; k < NDIM; ++k) acc += e[row * NDIM + k] * w[o * NDIM + k];
    m[i] = tanhf(3.0f * acc);
}

__global__ void k_outer(const float* __restrict__ m1, const float* __restrict__ m2,
                        float* __restrict__ a) {
    int idx = blockIdx.x * blockDim.x + threadIdx.x;
    if (idx >= NN * NN) return;
    int i = idx / NN, j = idx % NN;
    float d1 = 0.f, d2 = 0.f;
    for (int k = 0; k < NDIM; ++k) {
        d1 += m1[i * NDIM + k] * m2[j * NDIM + k];
        d2 += m2[i * NDIM + k] * m1[j * NDIM + k];
    }
    float v = tanhf(3.0f * (d1 - d2));
    a[idx] = v > 0.f ? v : 0.f;
}

// per-row top-40 mask (iterative argmax), one block per row
__global__ __launch_bounds__(128) void k_topk(const float* __restrict__ a,
                                              float* __restrict__ am) {
    __shared__ float v[NN];
    __shared__ unsigned char keep[NN];
    __shared__ float rv[128];
    __shared__ int   ri[128];
    const int row = blockIdx.x, tid = threadIdx.x;
    for (int j = tid; j < NN; j += 128) { v[j] = a[row * NN + j]; keep[j] = 0; }
    __syncthreads();
    for (int it = 0; it < TOPK; ++it) {
        float bv = -1e30f; int bi = -1;
        for (int j = tid; j < NN; j += 128)
            if (v[j] > bv) { bv = v[j]; bi = j; }
        rv[tid] = bv; ri[tid] = bi;
        __syncthreads();
        for (int s = 64; s > 0; s >>= 1) {
            if (tid < s && rv[tid + s] > rv[tid]) { rv[tid] = rv[tid + s]; ri[tid] = ri[tid + s]; }
            __syncthreads();
        }
        if (tid == 0) { keep[ri[0]] = 1; v[ri[0]] = -1e30f; }
        __syncthreads();
    }
    for (int j = tid; j < NN; j += 128)
        am[row * NN + j] = keep[j] ? a[row * NN + j] : 0.0f;
}

// A1 = rownorm(am + I) ; A2 = rownorm(am^T + I) ; emit bf16, one block per row
__global__ __launch_bounds__(128) void k_adjnorm(const float* __restrict__ am,
                                                 __bf16* __restrict__ A1,
                                                 __bf16* __restrict__ A2) {
    __shared__ float red[128];
    const int i = blockIdx.x, tid = threadIdx.x;
    float s1 = 0.f, s2 = 0.f;
    for (int j = tid; j < NN; j += 128) { s1 += am[i * NN + j]; s2 += am[j * NN + i]; }
    red[tid] = s1; __syncthreads();
    for (int s = 64; s > 0; s >>= 1) { if (tid < s) red[tid] += red[tid + s]; __syncthreads(); }
    s1 = red[0] + 1.0f; __syncthreads();
    red[tid] = s2; __syncthreads();
    for (int s = 64; s > 0; s >>= 1) { if (tid < s) red[tid] += red[tid + s]; __syncthreads(); }
    s2 = red[0] + 1.0f; __syncthreads();
    for (int j = tid; j < NN; j += 128) {
        float d = (i == j) ? 1.0f : 0.0f;
        A1[i * NN + j] = (__bf16)((am[i * NN + j] + d) / s1);
        A2[i * NN + j] = (__bf16)((am[j * NN + i] + d) / s2);
    }
}

// ------------------------------------------------- dilated inception + gating
struct GateW { const float* fw[4]; const float* fb[4];
               const float* gw[4]; const float* gb[4]; };

__global__ __launch_bounds__(256) void k_gate(const float* __restrict__ x, GateW W,
                                              float* __restrict__ xg_std,
                                              float* __restrict__ xg_nm,
                                              int Lin, int Lout) {
    __shared__ float lx[CR * 19];
    const int b = blockIdx.x / NN, n = blockIdx.x % NN;
    const int tid = threadIdx.x;
    for (int i = tid; i < CR * Lin; i += 256) {
        int c = i / Lin, t = i % Lin;
        lx[i] = x[(((size_t)(b * CR + c) * NN) + n) * Lin + t];
    }
    __syncthreads();
    const int kk[4] = {2, 3, 6, 7};
    const size_t M = (size_t)BB * CR * Lout;
    for (int i = tid; i < CR * Lout; i += 256) {
        int co = i / Lout, t = i % Lout;
        int br = co >> 3, cobr = co & 7, k = kk[br];
        const float* fw = W.fw[br]; const float* gw = W.gw[br];
        float fa = W.fb[br][cobr], ga = W.gb[br][cobr];
        int t0 = t + 7 - k;
        for (int ci = 0; ci < CR; ++ci) {
            const float* lr = &lx[ci * Lin + t0];
            const float* fwp = &fw[(cobr * CR + ci) * k];
            const float* gwp = &gw[(cobr * CR + ci) * k];
            for (int dt = 0; dt < k; ++dt) {
                float xv = lr[dt];
                fa += fwp[dt] * xv;
                ga += gwp[dt] * xv;
            }
        }
        float val = tanhf(fa) * (1.0f / (1.0f + expf(-ga)));
        xg_std[(((size_t)(b * CR + co) * NN) + n) * Lout + t] = val;
        xg_nm[(size_t)n * M + (size_t)(b * CR + co) * Lout + t] = val;
    }
}

// ---------------------------------------------------------------- pack kernel
// node-major f32 [512, M] -> K-blocked bf16 Hb[nblk][m][n&31]
// grid: (NN/32, M/64), block 256. LDS transpose (+1 pad, no bank conflicts),
// fully coalesced b128 stores.
__global__ __launch_bounds__(256) void k_pack(const float* __restrict__ Hf,
                                              __bf16* __restrict__ Hb, int M) {
    __shared__ float t[32 * 65];
    const int nb = blockIdx.x, m0 = blockIdx.y * 64;
    for (int i = threadIdx.x; i < 32 * 64; i += 256) {
        int r = i >> 6, c = i & 63;
        t[r * 65 + c] = Hf[(size_t)(nb * 32 + r) * M + m0 + c];
    }
    __syncthreads();
    const int ml = threadIdx.x >> 2, oct = threadIdx.x & 3;
    bf16x8 v;
    #pragma unroll
    for (int i = 0; i < 8; ++i) v[i] = (__bf16)t[(oct * 8 + i) * 65 + ml];
    *reinterpret_cast<bf16x8*>(Hb + ((size_t)nb * M + m0 + ml) * 32 + oct * 8) = v;
}

// ---------------------------------------------------------- skip convolutions
__global__ __launch_bounds__(64) void k_skipconv(const float* __restrict__ xg,
                                                 const float* __restrict__ w,
                                                 const float* __restrict__ bias,
                                                 float* __restrict__ skipbuf, int Lout) {
    __shared__ float lx[CR * 13];
    const int b = blockIdx.x / NN, n = blockIdx.x % NN;
    const int tid = threadIdx.x;
    for (int i = tid; i < CR * Lout; i += 64) {
        int c = i / Lout, t = i % Lout;
        lx[i] = xg[(((size_t)(b * CR + c) * NN) + n) * Lout + t];
    }
    __syncthreads();
    int cs = tid;
    float acc = bias[cs];
    for (int c = 0; c < CR; ++c)
        for (int l = 0; l < Lout; ++l)
            acc += w[(cs * CR + c) * Lout + l] * lx[c * Lout + l];
    skipbuf[((size_t)(b * CS + cs) * NN) + n] += acc;
}

__global__ __launch_bounds__(64) void k_skipE_relu(const float* __restrict__ skipbuf,
                                                   const float* __restrict__ xf,
                                                   const float* __restrict__ w,
                                                   const float* __restrict__ bias,
                                                   float* __restrict__ out) {
    const int b = blockIdx.x / NN, n = blockIdx.x % NN;
    const int cs = threadIdx.x;
    float acc = skipbuf[((size_t)(b * CS + cs) * NN) + n] + bias[cs];
    for (int c = 0; c < CR; ++c)
        acc += w[cs * CR + c] * xf[((size_t)(b * CR + c) * NN) + n];
    out[((size_t)(b * CS + cs) * NN) + n] = acc > 0.f ? acc : 0.f;
}

// --------------------------------------------------------- WMMA mixprop GEMM
// O[n,m] = 0.05*X[n,m] + 0.95 * sum_w A[n,w]*Hb[w,m]
// Wave computes a 16x64 strip: 1 A-frag x 4 B-frags / 4 accumulators.
// Block = 8 waves -> 32 rows x 256 cols. Inner k-step: 2 b128 (A) +
// 4 x 2 b128 (B) + 4 WMMA. Epilogue: 4 x one b128 blocked-bf16 store.
__global__ __launch_bounds__(256) void k_gemm_hop(const __bf16* __restrict__ A,
                                                  const __bf16* __restrict__ Hb,
                                                  const float*  __restrict__ X,
                                                  float* __restrict__ Of,
                                                  __bf16* __restrict__ Ob, int M) {
    const int lane = threadIdx.x & 31;
    const int wave = threadIdx.x >> 5;
    const int half = lane >> 4;
    const int l16  = lane & 15;
    const int rowBase = blockIdx.x * 32 + (wave >> 2) * 16;
    const int colBase = blockIdx.y * 256 + (wave & 3) * 64;
    const int mrow = rowBase + l16;          // A-frag row for this lane
    const __bf16* ap = A + (size_t)mrow * NN;
    const __bf16* hp = Hb + (size_t)(colBase + l16) * 32 + half * 8;
    f32x8 acc0 = {}, acc1 = {}, acc2 = {}, acc3 = {};
    #pragma unroll 2
    for (int k0 = 0; k0 < NN; k0 += 32) {
        // A frag (16x32): lane row = l16; K split per ISA 7.12.2 -> 2 x b128
        bf16x8 a0 = *reinterpret_cast<const bf16x8*>(ap + k0 + half * 8);
        bf16x8 a1 = *reinterpret_cast<const bf16x8*>(ap + k0 + 16 + half * 8);
        bf16x16 af;
        #pragma unroll
        for (int j = 0; j < 8; ++j) { af[j] = a0[j]; af[8 + j] = a1[j]; }
        const __bf16* hk = hp + (size_t)(k0 >> 5) * M * 32;
        #pragma unroll
        for (int u = 0; u < 4; ++u) {
            // B frag u (32x16): K = half*8 + j and 16 + half*8 + j -> 2 x b128
            const bf16x8* bp = reinterpret_cast<const bf16x8*>(hk + (size_t)u * 16 * 32);
            bf16x8 b0 = bp[0];
            bf16x8 b1 = bp[2];
            bf16x16 bf;
            #pragma unroll
            for (int j = 0; j < 8; ++j) { bf[j] = b0[j]; bf[8 + j] = b1[j]; }
            f32x8 c = (u == 0) ? acc0 : (u == 1) ? acc1 : (u == 2) ? acc2 : acc3;
            c = __builtin_amdgcn_wmma_f32_16x16x32_bf16(false, af, false, bf,
                                                        (short)0, c, false, false);
            if (u == 0) acc0 = c; else if (u == 1) acc1 = c;
            else if (u == 2) acc2 = c; else acc3 = c;
        }
    }
    #pragma unroll
    for (int u = 0; u < 4; ++u) {
        f32x8 c = (u == 0) ? acc0 : (u == 1) ? acc1 : (u == 2) ? acc2 : acc3;
        const int col = colBase + u * 16 + l16;
        float vals[8];
        #pragma unroll
        for (int r = 0; r < 8; ++r) {
            int row = rowBase + half * 8 + r;
            size_t idx = (size_t)row * M + col;
            float v = PROP_A * X[idx] + (1.0f - PROP_A) * c[r];
            Of[idx] = v;
            vals[r] = v;
        }
        if (Ob) {
            // rows rowBase+half*8 .. +7 contiguous in (n&31) -> one b128 store
            bf16x8 ov;
            #pragma unroll
            for (int r = 0; r < 8; ++r) ov[r] = (__bf16)vals[r];
            size_t bidx = ((size_t)blockIdx.x * M + col) * 32 + (rowBase & 31) + half * 8;
            *reinterpret_cast<bf16x8*>(Ob + bidx) = ov;
        }
    }
}

// -------------------------------------------- mixprop 1x1 mlp (+residual add)
__global__ __launch_bounds__(256) void k_mlp(const float* __restrict__ xprev,
                                             const float* __restrict__ h0,
                                             const float* __restrict__ h1a,
                                             const float* __restrict__ h2a,
                                             const float* __restrict__ h1b,
                                             const float* __restrict__ h2b,
                                             const float* __restrict__ g1w,
                                             const float* __restrict__ g1b,
                                             const float* __restrict__ g2w,
                                             const float* __restrict__ g2b,
                                             float* __restrict__ xnext,
                                             int Lin, int Lout) {
    __shared__ float ha[3 * CR * 13];
    __shared__ float hb[2 * CR * 13];
    const int b = blockIdx.x / NN, n = blockIdx.x % NN;
    const int tid = threadIdx.x;
    const int CL = CR * Lout;
    const size_t M = (size_t)BB * CL;
    const size_t base = (size_t)n * M + (size_t)b * CL;
    for (int i = tid; i < CL; i += 256) {
        ha[0 * CL + i] = h0[base + i];
        ha[1 * CL + i] = h1a[base + i];
        ha[2 * CL + i] = h2a[base + i];
        hb[0 * CL + i] = h1b[base + i];
        hb[1 * CL + i] = h2b[base + i];
    }
    __syncthreads();
    for (int i = tid; i < CL; i += 256) {
        int co = i / Lout, l = i % Lout;
        float acc = g1b[co] + g2b[co];
        for (int c = 0; c < CR; ++c) {
            float x0 = ha[c * Lout + l];
            acc += g1w[co * 96 +      c] * x0;
            acc += g1w[co * 96 + 32 + c] * ha[1 * CL + c * Lout + l];
            acc += g1w[co * 96 + 64 + c] * ha[2 * CL + c * Lout + l];
            acc += g2w[co * 96 +      c] * x0;
            acc += g2w[co * 96 + 32 + c] * hb[0 * CL + c * Lout + l];
            acc += g2w[co * 96 + 64 + c] * hb[1 * CL + c * Lout + l];
        }
        acc += xprev[(((size_t)(b * CR + co) * NN) + n) * Lin + (Lin - Lout) + l];
        xnext[(((size_t)(b * CR + co) * NN) + n) * Lout + l] = acc;
    }
}

// ----------------------------------------------------------------- layernorm
__global__ __launch_bounds__(256) void k_ln_reduce(const float* __restrict__ x,
                                                   float* __restrict__ sums, int nb) {
    __shared__ float rs[256], rq[256];
    const int b = blockIdx.x;
    const float* p = x + (size_t)b * nb;
    float s = 0.f, q = 0.f;
    for (int i = blockIdx.y * 256 + threadIdx.x; i < nb; i += gridDim.y * 256) {
        float v = p[i]; s += v; q += v * v;
    }
    rs[threadIdx.x] = s; rq[threadIdx.x] = q;
    __syncthreads();
    for (int st = 128; st > 0; st >>= 1) {
        if (threadIdx.x < st) { rs[threadIdx.x] += rs[threadIdx.x + st];
                                rq[threadIdx.x] += rq[threadIdx.x + st]; }
        __syncthreads();
    }
    if (threadIdx.x == 0) {
        atomicAdd(&sums[2 * b],     rs[0]);
        atomicAdd(&sums[2 * b + 1], rq[0]);
    }
}

__global__ void k_ln_apply(float* __restrict__ x, const float* __restrict__ sums,
                           const float* __restrict__ lnw, const float* __restrict__ lnb,
                           int nb, int total) {
    int i = blockIdx.x * blockDim.x + threadIdx.x;
    if (i >= total) return;
    int b = i / nb, w = i % nb;
    float inv_n = 1.0f / (float)nb;
    float mu  = sums[2 * b] * inv_n;
    float var = sums[2 * b + 1] * inv_n - mu * mu;
    x[i] = (x[i] - mu) * rsqrtf(var + EPSF) * lnw[w] + lnb[w];
}

// ===========================================================================
extern "C" void kernel_launch(void* const* d_in, const int* in_sizes, int n_in,
                              void* d_out, int out_size, void* d_ws, size_t ws_size,
                              hipStream_t stream) {
    (void)in_sizes; (void)n_in; (void)out_size; (void)ws_size;
    const float* x_in    = (const float*)d_in[0];
    const float* skip_in = (const float*)d_in[1];
    const float* emb1 = (const float*)d_in[2];
    const float* emb2 = (const float*)d_in[3];
    const float* l1w = (const float*)d_in[4]; const float* l1b = (const float*)d_in[5];
    const float* l2w = (const float*)d_in[6]; const float* l2b = (const float*)d_in[7];
    const float* sew = (const float*)d_in[8]; const float* seb = (const float*)d_in[9];

    size_t off = 0;
    auto carve = [&](size_t bytes) -> void* {
        void* p = (char*)d_ws + off;
        off += (bytes + 255) & ~(size_t)255;
        return p;
    };
    const size_t S19 = (size_t)BB * CR * NN * 19;
    const size_t S13 = (size_t)BB * CR * NN * 13;
    float* xA    = (float*)carve(S19 * 4);
    float* xB    = (float*)carve(S13 * 4);
    float* xg    = (float*)carve(S13 * 4);
    float* xgnm  = (float*)carve(S13 * 4);
    float* h1a   = (float*)carve(S13 * 4);
    float* h2a   = (float*)carve(S13 * 4);
    float* h1b   = (float*)carve(S13 * 4);
    float* h2b   = (float*)carve(S13 * 4);
    __bf16* xgnmb = (__bf16*)carve(S13 * 2);   // K-blocked
    __bf16* h1ab  = (__bf16*)carve(S13 * 2);   // K-blocked
    __bf16* h1bb  = (__bf16*)carve(S13 * 2);   // K-blocked
    float* skipbuf = (float*)carve((size_t)BB * CS * NN * 4);
    float* m1 = (float*)carve(NN * NDIM * 4);
    float* m2 = (float*)carve(NN * NDIM * 4);
    float* a_raw = (float*)carve((size_t)NN * NN * 4);
    float* a_m   = (float*)carve((size_t)NN * NN * 4);
    __bf16* A1 = (__bf16*)carve((size_t)NN * NN * 2);
    __bf16* A2 = (__bf16*)carve((size_t)NN * NN * 2);
    float* lnsums = (float*)carve(2 * BB * 4);

    // ---- graph construction
    {
        int t = NN * NDIM;
        k_mlin<<<(t + 255) / 256, 256, 0, stream>>>(emb1, l1w, l1b, m1);
        k_mlin<<<(t + 255) / 256, 256, 0, stream>>>(emb2, l2w, l2b, m2);
        k_outer<<<(NN * NN + 255) / 256, 256, 0, stream>>>(m1, m2, a_raw);
        k_topk<<<NN, 128, 0, stream>>>(a_raw, a_m);
        k_adjnorm<<<NN, 128, 0, stream>>>(a_m, A1, A2);
    }

    // ---- init skip accumulator & pad input to receptive field 19
    {
        int nsk = BB * CS * NN;
        k_copy<<<(nsk + 255) / 256, 256, 0, stream>>>(skip_in, skipbuf, nsk);
        k_pad<<<((int)S19 + 255) / 256, 256, 0, stream>>>(x_in, xA, (int)S19);
    }

    // ---- 3 temporal-graph layers
    const int Lins[3]  = {19, 13, 7};
    const int Louts[3] = {13, 7, 1};
    float* xcur = xA;
    float* xnxt = xB;
    for (int j = 0; j < 3; ++j) {
        const int Lin = Lins[j], Lout = Louts[j];
        const int base = 10 + 24 * j;
        GateW W;
        for (int q = 0; q < 4; ++q) {
            W.fw[q] = (const float*)d_in[base + q];
            W.fb[q] = (const float*)d_in[base + 4 + q];
            W.gw[q] = (const float*)d_in[base + 8 + q];
            W.gb[q] = (const float*)d_in[base + 12 + q];
        }
        const float* skw = (const float*)d_in[base + 16];
        const float* skb = (const float*)d_in[base + 17];
        const float* g1w = (const float*)d_in[base + 18];
        const float* g1b = (const float*)d_in[base + 19];
        const float* g2w = (const float*)d_in[base + 20];
        const float* g2b = (const float*)d_in[base + 21];
        const float* lnw = (const float*)d_in[base + 22];
        const float* lnb = (const float*)d_in[base + 23];

        const int M = BB * CR * Lout;               // 26624 / 14336 / 2048
        dim3 gpack(NN / 32, M / 64);
        dim3 ggemm(NN / 32, M / 256);

        // dilated inception + tanh*sigmoid gate
        k_gate<<<BB * NN, 256, 0, stream>>>(xcur, W, xg, xgnm, Lin, Lout);
        // pack node-major f32 -> K-blocked bf16 for GEMM B operand
        k_pack<<<gpack, 256, 0, stream>>>(xgnm, xgnmb, M);
        // skip accumulation
        k_skipconv<<<BB * NN, 64, 0, stream>>>(xg, skw, skb, skipbuf, Lout);

        // mixprop hops: 4 WMMA GEMMs (A1 path, A2 path, 2 hops each)
        k_gemm_hop<<<ggemm, 256, 0, stream>>>(A1, xgnmb, xgnm, h1a, h1ab, M);
        k_gemm_hop<<<ggemm, 256, 0, stream>>>(A1, h1ab,  xgnm, h2a, (__bf16*)nullptr, M);
        k_gemm_hop<<<ggemm, 256, 0, stream>>>(A2, xgnmb, xgnm, h1b, h1bb, M);
        k_gemm_hop<<<ggemm, 256, 0, stream>>>(A2, h1bb,  xgnm, h2b, (__bf16*)nullptr, M);

        // concat-hop 1x1 conv (both paths) + residual
        k_mlp<<<BB * NN, 256, 0, stream>>>(xcur, xgnm, h1a, h2a, h1b, h2b,
                                           g1w, g1b, g2w, g2b, xnxt, Lin, Lout);

        // layer norm over (C,N,L) per batch
        const int nb = CR * NN * Lout;
        hipMemsetAsync(lnsums, 0, 2 * BB * sizeof(float), stream);
        dim3 rg(BB, 32);
        k_ln_reduce<<<rg, 256, 0, stream>>>(xnxt, lnsums, nb);
        int total = BB * nb;
        k_ln_apply<<<(total + 255) / 256, 256, 0, stream>>>(xnxt, lnsums, lnw, lnb, nb, total);

        float* t = xcur; xcur = xnxt; xnxt = t;   // ping-pong
    }

    // ---- final skip conv (1x1 on L=1 output) + ReLU -> d_out [B,64,N,1]
    k_skipE_relu<<<BB * NN, 64, 0, stream>>>(skipbuf, xcur, sew, seb, (float*)d_out);
}
